// Decoder_39298950758847
// MI455X (gfx1250) — matrix operand order repaired and checked by reading, not compiled, not run on hardware
//
#include <hip/hip_runtime.h>
#include <hip/hip_bf16.h>

typedef _Float16 v16h __attribute__((ext_vector_type(16)));
typedef float    v8f  __attribute__((ext_vector_type(8)));

__device__ __forceinline__ v8f wmma16(v16h a, v16h b, v8f c) {
    // D = A(16x32 f16) x B(32x16 f16) + C(16x16 f32)
    return __builtin_amdgcn_wmma_f32_16x16x32_f16(false, a, false, b, (short)0, c, false, false);
}

// A-operand (16x32 f16) for this lane from an LDS row-major tile (two 16B chunks).
// a[i] <- k = kbase + (i>=8 ? 16 : 0) + (i&7), with kbase = 32*kk + laneHalf*8
__device__ __forceinline__ v16h loadA(const _Float16* rowPtr, int kbase) {
    union { v16h v; uint4 q[2]; } u;
    u.q[0] = *(const uint4*)(rowPtr + kbase);
    u.q[1] = *(const uint4*)(rowPtr + kbase + 16);
    return u.v;
}

// B-operand (32x16 f16): per lane two 16B chunks of K for a fixed column.
// b[i] <- k = k0 + (i>=8 ? 16 : 0) + (i&7), with k0 = 32*kk + laneHalf*8
__device__ __forceinline__ v16h loadB(const _Float16* p) {
    union { v16h v; uint4 q[2]; } u;
    u.q[0] = *(const uint4*)(p);
    u.q[1] = *(const uint4*)(p + 16);
    return u.v;
}

__device__ __forceinline__ float fast_silu(float x) {
    // x * 1/(1+e^-x) with v_rcp_f32 instead of IEEE divide
    return x * __builtin_amdgcn_rcpf(1.0f + __expf(-x));
}

// ---------------------------------------------------------------------------
// Node kernel: 4 MLPs (128->128 SiLU ->{2,2,1,3}) per node, then 2x2 algebra.
// Layer1: D = W1^T x X^T (weights = A in VGPRs, X = B from LDS).
// Layer2: D = W2cat^T x H^T (both operands from LDS), one MLP per wave 0..3.
// ---------------------------------------------------------------------------
__global__ __launch_bounds__(256, 1)
void node_kernel(const float* __restrict__ x,
                 const float* __restrict__ EgW1, const float* __restrict__ Egb1,
                 const float* __restrict__ EgW2, const float* __restrict__ Egb2,
                 const float* __restrict__ SgW1, const float* __restrict__ Sgb1,
                 const float* __restrict__ SgW2, const float* __restrict__ Sgb2,
                 const float* __restrict__ LnW1, const float* __restrict__ Lnb1,
                 const float* __restrict__ LnW2, const float* __restrict__ Lnb2,
                 const float* __restrict__ MnW1, const float* __restrict__ Mnb1,
                 const float* __restrict__ MnW2, const float* __restrict__ Mnb2,
                 float* __restrict__ out, float* __restrict__ ws,
                 int N, int ntiles) {
    __shared__ __align__(16) _Float16 sX[16 * 128];       // [node][feat] f16
    __shared__ __align__(16) _Float16 sH[4 * 16 * 128];   // [mlp][node][k] f16 (row-major!)
    __shared__ __align__(16) _Float16 sW2T[4 * 16 * 128]; // [mlp][o(pad16)][k] f16
    __shared__ float sOut[16 * 8];

    const int tid      = threadIdx.x;
    const int lane     = tid & 31;
    const int wave     = tid >> 5;          // 0..7 -> layer1 out cols [16w,16w+16)
    const int laneHalf = lane >> 4;         // 0/1
    const int l15      = lane & 15;

    // ---- one-time LDS init: W2^T cat, padded to 16 rows per MLP ----
    for (int idx = tid; idx < 4 * 16 * 128; idx += 256) {
        int m = idx >> 11, o = (idx >> 7) & 15, k = idx & 127;
        float wv = 0.0f;
        if (m == 0)      { if (o < 2) wv = EgW2[k * 2 + o]; }
        else if (m == 1) { if (o < 2) wv = SgW2[k * 2 + o]; }
        else if (m == 2) { if (o < 1) wv = LnW2[k]; }
        else             { if (o < 3) wv = MnW2[k * 3 + o]; }
        sW2T[idx] = (_Float16)wv;
    }

    // ---- layer-1 weights as A-operands in VGPRs (row M = out col, K contiguous-ish) ----
    v16h aReg[4][4];
    float biasReg[4][8];
    {
        const int col = (wave << 4) + l15;   // A row -> global out column
        #pragma unroll
        for (int m = 0; m < 4; ++m) {
            const float* W = (m == 0) ? EgW1 : (m == 1) ? SgW1 : (m == 2) ? LnW1 : MnW1;
            const float* B = (m == 0) ? Egb1 : (m == 1) ? Sgb1 : (m == 2) ? Lnb1 : Mnb1;
            #pragma unroll
            for (int kk = 0; kk < 4; ++kk) {
                union { v16h v; _Float16 h[16]; } u;
                #pragma unroll
                for (int i = 0; i < 16; ++i) {
                    int k = kk * 32 + ((i >> 3) << 4) + laneHalf * 8 + (i & 7);
                    u.h[i] = (_Float16)W[k * 128 + col];
                }
                aReg[m][kk] = u.v;
            }
            #pragma unroll
            for (int r = 0; r < 8; ++r)
                biasReg[m][r] = B[(wave << 4) + laneHalf * 8 + r];
        }
    }

    // layer-2 per-wave constants (wave m handles MLP m)
    const float* b2p = Egb2; int obase = 0, ocnt = 2;
    if (wave == 1) { b2p = Sgb2; obase = 2; ocnt = 2; }
    else if (wave == 2) { b2p = Lnb2; obase = 4; ocnt = 1; }
    else if (wave == 3) { b2p = Mnb2; obase = 5; ocnt = 3; }
    float b2v[3];
    #pragma unroll
    for (int r = 0; r < 3; ++r) b2v[r] = (r < ocnt) ? b2p[r] : 0.0f;

    for (int tile = blockIdx.x; tile < ntiles; tile += gridDim.x) {
        // stage 16x128 input tile as f16 (vectorized, branchless guard)
        #pragma unroll
        for (int it = 0; it < 2; ++it) {
            int idx = tid + it * 256;                    // 0..511 float4 chunks
            int row = idx >> 5, c4 = (idx & 31) << 2;
            int nd = tile * 16 + row;
            int ndc = nd < N ? nd : 0;
            float4 f = *(const float4*)(x + (size_t)ndc * 128 + c4);
            if (nd >= N) f = make_float4(0.f, 0.f, 0.f, 0.f);
            union { _Float16 h[4]; uint2 q; } p;
            p.h[0] = (_Float16)f.x; p.h[1] = (_Float16)f.y;
            p.h[2] = (_Float16)f.z; p.h[3] = (_Float16)f.w;
            *(uint2*)(sX + row * 128 + c4) = p.q;
        }
        __syncthreads();

        // ---- layer 1: acc[m] = W1_m^T x X^T over 4 K-steps ----
        v8f acc[4];
        #pragma unroll
        for (int m = 0; m < 4; ++m) acc[m] = (v8f){0.f,0.f,0.f,0.f,0.f,0.f,0.f,0.f};
        const _Float16* xrow = sX + l15 * 128;           // B column = node l15
        #pragma unroll
        for (int kk = 0; kk < 4; ++kk) {
            v16h b = loadB(xrow + kk * 32 + laneHalf * 8);
            #pragma unroll
            for (int m = 0; m < 4; ++m) acc[m] = wmma16(aReg[m][kk], b, acc[m]);
        }
        // D lane layout: node = l15, colLocal = laneHalf*8 + r  -> row-major b128 store
        #pragma unroll
        for (int m = 0; m < 4; ++m) {
            union { uint4 q; _Float16 h[8]; } p;
            #pragma unroll
            for (int r = 0; r < 8; ++r)
                p.h[r] = (_Float16)fast_silu(acc[m][r] + biasReg[m][r]);
            *(uint4*)(sH + m * 2048 + l15 * 128 + (wave << 4) + laneHalf * 8) = p.q;
        }
        __syncthreads();

        // ---- layer 2: wave m computes out_m = W2_m^T x H_m^T (4 WMMAs) ----
        if (wave < 4) {
            const int m = wave;
            v8f acc2 = (v8f){0.f,0.f,0.f,0.f,0.f,0.f,0.f,0.f};
            const _Float16* w2row = sW2T + m * 2048 + l15 * 128;  // A row = o
            const _Float16* hrow  = sH   + m * 2048 + l15 * 128;  // B col = node
            #pragma unroll
            for (int kk = 0; kk < 4; ++kk) {
                v16h a2 = loadA(w2row, kk * 32 + laneHalf * 8);
                v16h b2 = loadB(hrow + kk * 32 + laneHalf * 8);
                acc2 = wmma16(a2, b2, acc2);
            }
            // lanes 0..15 (laneHalf==0): VGPR r holds o=r for node=lane
            if (lane < 16) {
                #pragma unroll
                for (int r = 0; r < 3; ++r)
                    if (r < ocnt) sOut[lane * 8 + obase + r] = acc2[r] + b2v[r];
            }
        }
        __syncthreads();

        // ---- finals: L = [[0,-l],[l,0]]; M = Mm Mm^T, Mm = [[m0,0],[m1,m2]] ----
        if (tid < 16) {
            int nd = tile * 16 + tid;
            if (nd < N) {
                const float* o8 = sOut + tid * 8;
                float e0 = o8[0], e1 = o8[1], s0 = o8[2], s1 = o8[3];
                float l = o8[4], m0 = o8[5], m1 = o8[6], m2 = o8[7];
                float M00 = m0 * m0, M01 = m0 * m1, M11 = m1 * m1 + m2 * m2;
                out[2 * nd + 0] = -l * e1 + M00 * s0 + M01 * s1;      // nodes_term
                out[2 * nd + 1] =  l * e0 + M01 * s0 + M11 * s1;
                out[2 * N + 2 * nd + 0] = M00 * e0 + M01 * e1;        // deg_E
                out[2 * N + 2 * nd + 1] = M01 * e0 + M11 * e1;
                out[4 * N + 2 * nd + 0] = -l * s1;                    // deg_S
                out[4 * N + 2 * nd + 1] =  l * s0;
                ws[2 * nd + 0] = e0; ws[2 * nd + 1] = e1;             // dEdz
                ws[2 * N + 2 * nd + 0] = s0; ws[2 * N + 2 * nd + 1] = s1; // dSdz
            }
        }
        __syncthreads();
    }
}

// ---------------------------------------------------------------------------
// Edge kernel: 2 MLPs (384->128 SiLU ->{1,3}), then scatter-add -edges_term.
// ---------------------------------------------------------------------------
__global__ __launch_bounds__(256, 1)
void edge_kernel(const float* __restrict__ node_attr,
                 const int* __restrict__ eidx,
                 const float* __restrict__ edge_attr,
                 const float* __restrict__ LeW1, const float* __restrict__ Leb1,
                 const float* __restrict__ LeW2, const float* __restrict__ Leb2,
                 const float* __restrict__ MeW1, const float* __restrict__ Meb1,
                 const float* __restrict__ MeW2, const float* __restrict__ Meb2,
                 const float* __restrict__ ws, float* __restrict__ out,
                 int N, int E, int ntiles) {
    __shared__ __align__(16) _Float16 sIn[16 * 384];      // [edge][3H] f16
    __shared__ __align__(16) _Float16 sH[2 * 16 * 128];   // [mlp][edge][k] f16 (row-major)
    __shared__ __align__(16) _Float16 sW2T[2 * 16 * 128]; // [mlp][o(pad16)][k] f16
    __shared__ float sOut[16 * 4];
    __shared__ int   sidx[32];                            // [0..15]=senders, [16..31]=receivers

    const int tid      = threadIdx.x;
    const int lane     = tid & 31;
    const int wave     = tid >> 5;
    const int laneHalf = lane >> 4;
    const int l15      = lane & 15;

    // ---- one-time LDS init: W2^T, padded rows ----
    for (int idx = tid; idx < 2 * 16 * 128; idx += 256) {
        int m = idx >> 11, o = (idx >> 7) & 15, k = idx & 127;
        float wv = 0.0f;
        if (m == 0) { if (o == 0) wv = LeW2[k]; }
        else        { if (o < 3)  wv = MeW2[k * 3 + o]; }
        sW2T[idx] = (_Float16)wv;
    }

    // ---- layer-1 weights as A-operands in VGPRs: 2 MLPs x 12 K-steps ----
    v16h aReg[2][12];
    float biasReg[2][8];
    {
        const int col = (wave << 4) + l15;
        #pragma unroll
        for (int m = 0; m < 2; ++m) {
            const float* W = m ? MeW1 : LeW1;   // (384,128) row-major
            const float* B = m ? Meb1 : Leb1;
            #pragma unroll
            for (int kk = 0; kk < 12; ++kk) {
                union { v16h v; _Float16 h[16]; } u;
                #pragma unroll
                for (int i = 0; i < 16; ++i) {
                    int k = kk * 32 + ((i >> 3) << 4) + laneHalf * 8 + (i & 7);
                    u.h[i] = (_Float16)W[k * 128 + col];
                }
                aReg[m][kk] = u.v;
            }
            #pragma unroll
            for (int r = 0; r < 8; ++r)
                biasReg[m][r] = B[(wave << 4) + laneHalf * 8 + r];
        }
    }

    // layer-2 per-wave constants (wave 0 -> Le, wave 1 -> Me)
    const float* b2p = Leb2; int obase = 0, ocnt = 1;
    if (wave == 1) { b2p = Meb2; obase = 1; ocnt = 3; }
    float b2v[3];
    #pragma unroll
    for (int r = 0; r < 3; ++r) b2v[r] = (r < ocnt) ? b2p[r] : 0.0f;

    for (int tile = blockIdx.x; tile < ntiles; tile += gridDim.x) {
        if (tile + gridDim.x < ntiles)
            __builtin_prefetch(edge_attr + (size_t)(tile + gridDim.x) * 16 * 128 + tid * 8, 0, 0);

        // stage edge_in = [edge_attr | x[snd] | x[rcv]] as f16 (float4 chunks)
        #pragma unroll
        for (int it = 0; it < 6; ++it) {
            int idx = tid + it * 256;                 // 0..1535
            int row = idx / 96, cc = idx - row * 96;
            int c4 = cc << 2;                         // 0..380, one 128-segment each
            int e = tile * 16 + row;
            int ec = e < E ? e : 0;
            const float* src;
            if (c4 < 128)      src = edge_attr + (size_t)ec * 128 + c4;
            else if (c4 < 256) { int s = eidx[ec];     src = node_attr + (size_t)s * 128 + (c4 - 128); }
            else               { int r = eidx[E + ec]; src = node_attr + (size_t)r * 128 + (c4 - 256); }
            float4 f = *(const float4*)src;
            if (e >= E) f = make_float4(0.f, 0.f, 0.f, 0.f);
            union { _Float16 h[4]; uint2 q; } p;
            p.h[0] = (_Float16)f.x; p.h[1] = (_Float16)f.y;
            p.h[2] = (_Float16)f.z; p.h[3] = (_Float16)f.w;
            *(uint2*)(sIn + row * 384 + c4) = p.q;
        }
        if (tid < 32) {
            int e = tile * 16 + (tid & 15);
            sidx[tid] = (e < E) ? eidx[(tid >> 4) * E + e] : 0;
        }
        __syncthreads();

        // ---- layer 1: 12 K-steps, 2 MLPs per wave ----
        v8f acc0 = (v8f){0.f,0.f,0.f,0.f,0.f,0.f,0.f,0.f};
        v8f acc1 = acc0;
        const _Float16* erow = sIn + l15 * 384;       // B column = edge l15
        #pragma unroll
        for (int kk = 0; kk < 12; ++kk) {
            v16h b = loadB(erow + kk * 32 + laneHalf * 8);
            acc0 = wmma16(aReg[0][kk], b, acc0);
            acc1 = wmma16(aReg[1][kk], b, acc1);
        }
        #pragma unroll
        for (int m = 0; m < 2; ++m) {
            v8f acc = m ? acc1 : acc0;
            union { uint4 q; _Float16 h[8]; } p;
            #pragma unroll
            for (int r = 0; r < 8; ++r)
                p.h[r] = (_Float16)fast_silu(acc[r] + biasReg[m][r]);
            *(uint4*)(sH + m * 2048 + l15 * 128 + (wave << 4) + laneHalf * 8) = p.q;
        }
        __syncthreads();

        // ---- layer 2: wave m computes W2_m^T x H_m^T (4 WMMAs) ----
        if (wave < 2) {
            const int m = wave;
            v8f acc2 = (v8f){0.f,0.f,0.f,0.f,0.f,0.f,0.f,0.f};
            const _Float16* w2row = sW2T + m * 2048 + l15 * 128;
            const _Float16* hrow  = sH   + m * 2048 + l15 * 128;
            #pragma unroll
            for (int kk = 0; kk < 4; ++kk) {
                v16h a2 = loadA(w2row, kk * 32 + laneHalf * 8);
                v16h b2 = loadB(hrow + kk * 32 + laneHalf * 8);
                acc2 = wmma16(a2, b2, acc2);
            }
            if (lane < 16) {
                #pragma unroll
                for (int r = 0; r < 3; ++r)
                    if (r < ocnt) sOut[lane * 4 + obase + r] = acc2[r] + b2v[r];
            }
        }
        __syncthreads();

        // ---- finals: edges_term = L_e @ dEdz[snd] + M_e @ dSdz[snd]; scatter -term ----
        if (tid < 16) {
            int e = tile * 16 + tid;
            if (e < E) {
                const float* o4 = sOut + tid * 4;
                float l = o4[0], m0 = o4[1], m1 = o4[2], m2 = o4[3];
                int snd = sidx[tid], rcv = sidx[16 + tid];
                float e0 = ws[2 * snd + 0], e1 = ws[2 * snd + 1];
                float s0 = ws[2 * N + 2 * snd + 0], s1 = ws[2 * N + 2 * snd + 1];
                float M00 = m0 * m0, M01 = m0 * m1, M11 = m1 * m1 + m2 * m2;
                float t0 = -l * e1 + M00 * s0 + M01 * s1;
                float t1 =  l * e0 + M01 * s0 + M11 * s1;
                atomicAdd(out + 2 * rcv + 0, -t0);
                atomicAdd(out + 2 * rcv + 1, -t1);
            }
        }
        __syncthreads();
    }
}

extern "C" void kernel_launch(void* const* d_in, const int* in_sizes, int n_in,
                              void* d_out, int out_size, void* d_ws, size_t ws_size,
                              hipStream_t stream) {
    const float* node_attr = (const float*)d_in[0];
    const int*   eidx      = (const int*)d_in[1];
    const float* edge_attr = (const float*)d_in[2];
    const float* EgW1 = (const float*)d_in[3];  const float* Egb1 = (const float*)d_in[4];
    const float* EgW2 = (const float*)d_in[5];  const float* Egb2 = (const float*)d_in[6];
    const float* SgW1 = (const float*)d_in[7];  const float* Sgb1 = (const float*)d_in[8];
    const float* SgW2 = (const float*)d_in[9];  const float* Sgb2 = (const float*)d_in[10];
    const float* LnW1 = (const float*)d_in[11]; const float* Lnb1 = (const float*)d_in[12];
    const float* LnW2 = (const float*)d_in[13]; const float* Lnb2 = (const float*)d_in[14];
    const float* MnW1 = (const float*)d_in[15]; const float* Mnb1 = (const float*)d_in[16];
    const float* MnW2 = (const float*)d_in[17]; const float* Mnb2 = (const float*)d_in[18];
    const float* LeW1 = (const float*)d_in[19]; const float* Leb1 = (const float*)d_in[20];
    const float* LeW2 = (const float*)d_in[21]; const float* Leb2 = (const float*)d_in[22];
    const float* MeW1 = (const float*)d_in[23]; const float* Meb1 = (const float*)d_in[24];
    const float* MeW2 = (const float*)d_in[25]; const float* Meb2 = (const float*)d_in[26];

    float* out = (float*)d_out;
    float* ws  = (float*)d_ws;

    const int N = in_sizes[0] / 128;
    const int E = in_sizes[2] / 128;
    const int ntN = (N + 15) / 16;
    const int ntE = (E + 15) / 16;
    const int gN = ntN < 1024 ? ntN : 1024;
    const int gE = ntE < 2048 ? ntE : 2048;

    node_kernel<<<gN, 256, 0, stream>>>(node_attr,
        EgW1, Egb1, EgW2, Egb2, SgW1, Sgb1, SgW2, Sgb2,
        LnW1, Lnb1, LnW2, Lnb2, MnW1, Mnb1, MnW2, Mnb2,
        out, ws, N, ntN);

    edge_kernel<<<gE, 256, 0, stream>>>(node_attr, eidx, edge_attr,
        LeW1, Leb1, LeW2, Leb2, MeW1, Meb1, MeW2, Meb2,
        ws, out, N, E, ntE);
}
